// KiloNeRF_88064009437975
// MI455X (gfx1250) — matrix operand order repaired and checked by reading, not compiled, not run on hardware
//
#include <hip/hip_runtime.h>

// KiloNeRF fused evaluation for MI455X (gfx1250, wave32, WMMA f16->f32).
// One block (8 waves) per tiny network; weights live in LDS as f16;
// x tiles are streamed into per-wave LDS buffers with async global->LDS
// B128 loads (ASYNCcnt), overlapping the next tile's copy with the current
// tile's 14 v_wmma_f32_16x16x32_f16 ops.

typedef __attribute__((ext_vector_type(16))) _Float16 v16h;
typedef __attribute__((ext_vector_type(8)))  float    v8f;
typedef int v4i __attribute__((vector_size(16)));   // type expected by async-LDS builtin

#define PC   63
#define DC   27
#define XDIM (PC + DC)   // 90
#define HID  32
#define PTS  512
#define NNET 4096

#if __has_builtin(__builtin_amdgcn_global_load_async_to_lds_b128)
#define HAS_ASYNC_LDS 1
#else
#define HAS_ASYNC_LDS 0
#warning "builtin global_load_async_to_lds_b128 not available: using synchronous LDS copy fallback"
#endif

static __device__ inline void wait_async0() {
#if __has_builtin(__builtin_amdgcn_s_wait_asynccnt)
    __builtin_amdgcn_s_wait_asynccnt(0);
    asm volatile("" ::: "memory");
#else
    asm volatile("s_wait_asynccnt 0x0" ::: "memory");
#endif
}

static __device__ inline void wait_ds0() {
    asm volatile("s_wait_dscnt 0x0" ::: "memory");
}

// Copy one 16-point x tile (16*90 f32 = 360 float4, contiguous) into LDS.
static __device__ inline void copy_tile_async(const float* g, float* l, int lane) {
#if HAS_ASYNC_LDS
#pragma unroll
    for (int i = 0; i < 12; ++i) {
        const int idx = lane + i * 32;
        if (idx < 360) {
            __builtin_amdgcn_global_load_async_to_lds_b128(
                (v4i*)(g + (size_t)idx * 4), (v4i*)(l + (size_t)idx * 4),
                /*offset=*/0, /*cpol=*/0);
        }
    }
#else
    for (int idx = lane; idx < 360; idx += 32) {
        ((float4*)l)[idx] = ((const float4*)g)[idx];
    }
    asm volatile("" ::: "memory");
#endif
}

static __device__ inline v8f wmma_f(v16h a, v16h b, v8f c) {
    // D = A(16x32 f16) * B(32x16 f16) + C(16x16 f32)
    return __builtin_amdgcn_wmma_f32_16x16x32_f16(
        /*neg_a=*/false, a, /*neg_b=*/false, b,
        /*c_mod=*/(short)0, c, /*reuse_a=*/false, /*reuse_b=*/false);
}

// B fragment (32x16) from LDS weight image W stored row-major [out][ld] (f16).
// Lane l: N = l&15 (+16*ntile), K = 2*(v&3) + 8*(l>>4) + 16*(v>>2) + kbase.
static __device__ inline v16h load_bfrag(const _Float16* W, int ld, int lane,
                                         int ntile, int kbase) {
    const int n  = (lane & 15) + (ntile << 4);
    const int kb = kbase + ((lane >> 4) << 3);
    const _Float16* row = W + n * ld + kb;
    v16h f;
#pragma unroll
    for (int v = 0; v < 8; ++v) {
        f[2 * v]     = row[((v & 3) << 1) + ((v >> 2) << 4)];
        f[2 * v + 1] = row[((v & 3) << 1) + ((v >> 2) << 4) + 1];
    }
    return f;
}

// A fragment (16x32) from LDS staging, row-major [16][32] f16.
// Lane l: M = l&15, K = 2*(v&3) + 8*(l>>4) + 16*(v>>2).
static __device__ inline v16h load_afrag_lds(const _Float16* S, int lane) {
    const int m  = lane & 15;
    const int kh = (lane >> 4) << 3;
    const _Float16* row = S + m * 32;
    v16h f;
#pragma unroll
    for (int v = 0; v < 8; ++v) {
        const int k = ((v & 3) << 1) + kh + ((v >> 2) << 4);
        f[2 * v]     = row[k];
        f[2 * v + 1] = row[k + 1];
    }
    return f;
}

// A fragment (16x32) from the LDS x-tile (f32 rows of length XDIM),
// columns [kofs, kofs+kmax) valid, rest zero-padded.
static __device__ inline v16h load_afrag_xlds(const float* X, int lane,
                                              int kofs, int kmax) {
    const int m  = lane & 15;
    const int kh = (lane >> 4) << 3;
    const float* row = X + m * XDIM;
    v16h f;
#pragma unroll
    for (int v = 0; v < 8; ++v) {
        const int k = ((v & 3) << 1) + kh + ((v >> 2) << 4);
        const float a = (k     < kmax) ? row[kofs + k]     : 0.0f;
        const float b = (k + 1 < kmax) ? row[kofs + k + 1] : 0.0f;
        f[2 * v]     = (_Float16)a;
        f[2 * v + 1] = (_Float16)b;
    }
    return f;
}

// Broadcast per-output-channel bias into C accumulator (N = lane&15).
static __device__ inline v8f bias_c(const float* B, int lane, int ntile) {
    const float bv = B[(ntile << 4) + (lane & 15)];
    v8f c;
#pragma unroll
    for (int i = 0; i < 8; ++i) c[i] = bv;
    return c;
}

static __device__ inline v8f relu8(v8f c) {
#pragma unroll
    for (int i = 0; i < 8; ++i) c[i] = fmaxf(c[i], 0.0f);
    return c;
}

// Store a 16x32 f32 D-tile (two 16x16 accumulators) to LDS staging as f16.
// Lane l holds (M = r + 8*(l>>4), N = l&15).
static __device__ inline void store_stage(_Float16* S, int lane,
                                          const v8f& c0, const v8f& c1) {
    const int n  = lane & 15;
    const int mh = (lane >> 4) << 3;
#pragma unroll
    for (int r = 0; r < 8; ++r) {
        S[(mh + r) * 32 + n]      = (_Float16)c0[r];
        S[(mh + r) * 32 + 16 + n] = (_Float16)c1[r];
    }
}

__global__ __launch_bounds__(256) void kilonerf_kernel(
    const float* __restrict__ x,
    const float* __restrict__ W0, const float* __restrict__ b0,
    const float* __restrict__ W1, const float* __restrict__ b1,
    const float* __restrict__ Wa, const float* __restrict__ ba,
    const float* __restrict__ Wf, const float* __restrict__ bf,
    const float* __restrict__ Wd, const float* __restrict__ bd,
    const float* __restrict__ Wr, const float* __restrict__ br,
    float* __restrict__ out)
{
    // ---- LDS: f16 weight images (zero padded) + f32 biases + staging ----
    __shared__ _Float16 sW0[32 * 64];   // [32 out][64 in]  (63 real)
    __shared__ _Float16 sW1[32 * 32];
    __shared__ _Float16 sWa[16 * 32];   // 1 real row
    __shared__ _Float16 sWf[32 * 32];
    __shared__ _Float16 sWd[32 * 64];   // 59 real in
    __shared__ _Float16 sWr[16 * 32];   // 3 real rows
    __shared__ float sB0[32], sB1[32], sBa[16], sBf[32], sBd[32], sBr[16];
    __shared__ __align__(16) _Float16 sStage[8][16 * 32];   // per-wave 1KB
    __shared__ __align__(16) float    sXbuf[8][16 * XDIM];  // per-wave x tile

    const int net  = blockIdx.x;
    const int tid  = threadIdx.x;
    const int lane = tid & 31;
    const int wid  = tid >> 5;

    // ---- zero-pad weight images ----
    for (int i = tid; i < 32 * 64; i += 256) { sW0[i] = (_Float16)0.0f; sWd[i] = (_Float16)0.0f; }
    for (int i = tid; i < 16 * 32; i += 256) { sWa[i] = (_Float16)0.0f; sWr[i] = (_Float16)0.0f; }
    __syncthreads();

    // ---- load + convert weights/biases for this network ----
    {
        const float* g = W0 + (size_t)net * 32 * PC;
        for (int i = tid; i < 32 * PC; i += 256) sW0[(i / PC) * 64 + (i % PC)] = (_Float16)g[i];
    }
    {
        const float* g = W1 + (size_t)net * 32 * 32;
        for (int i = tid; i < 32 * 32; i += 256) sW1[i] = (_Float16)g[i];
    }
    {
        const float* g = Wa + (size_t)net * 32;            // [1][32]
        for (int i = tid; i < 32; i += 256) sWa[i] = (_Float16)g[i];
    }
    {
        const float* g = Wf + (size_t)net * 32 * 32;
        for (int i = tid; i < 32 * 32; i += 256) sWf[i] = (_Float16)g[i];
    }
    {
        const float* g = Wd + (size_t)net * 32 * 59;       // [32][59]
        for (int i = tid; i < 32 * 59; i += 256) sWd[(i / 59) * 64 + (i % 59)] = (_Float16)g[i];
    }
    {
        const float* g = Wr + (size_t)net * 3 * 32;        // [3][32]
        for (int i = tid; i < 3 * 32; i += 256) sWr[i] = (_Float16)g[i];
    }
    if (tid < 32) {
        sB0[tid] = b0[(size_t)net * 32 + tid];
        sB1[tid] = b1[(size_t)net * 32 + tid];
        sBf[tid] = bf[(size_t)net * 32 + tid];
        sBd[tid] = bd[(size_t)net * 32 + tid];
    }
    if (tid < 16) {
        sBa[tid] = (tid == 0) ? ba[net] : 0.0f;
        sBr[tid] = (tid < 3) ? br[(size_t)net * 3 + tid] : 0.0f;
    }
    __syncthreads();

    const float* xnet = x   + (size_t)net * PTS * XDIM;
    float*       onet = out + (size_t)net * PTS * 4;
    _Float16*    S    = &sStage[wid][0];
    float*       XB   = &sXbuf[wid][0];

    // kick off async copy of this wave's first x tile
    copy_tile_async(xnet + (size_t)(wid * 4) * 16 * XDIM, XB, lane);

    // ---- 8 waves x 4 tiles x 16 points = 512 points ----
    for (int tt = 0; tt < 4; ++tt) {
        const int m0 = (wid * 4 + tt) * 16;

        // x tile is resident; pull all three A fragments now
        wait_async0();
        v16h a0 = load_afrag_xlds(XB, lane, 0, 32);     // pos K 0..31
        v16h a1 = load_afrag_xlds(XB, lane, 32, 31);    // pos K 32..62, pad 63
        v16h ad = load_afrag_xlds(XB, lane, PC, DC);    // dir K 0..26, pad
        wait_ds0();                                      // frags landed in VGPRs

        // overlap: async-refill the buffer with the next tile
        if (tt < 3)
            copy_tile_async(xnet + (size_t)(m0 + 16) * XDIM, XB, lane);

        // Layer 0: [16x63] x [63x32], K padded to 64
        v8f c0 = bias_c(sB0, lane, 0);
        v8f c1 = bias_c(sB0, lane, 1);
        c0 = wmma_f(a0, load_bfrag(sW0, 64, lane, 0, 0),  c0);
        c0 = wmma_f(a1, load_bfrag(sW0, 64, lane, 0, 32), c0);
        c1 = wmma_f(a0, load_bfrag(sW0, 64, lane, 1, 0),  c1);
        c1 = wmma_f(a1, load_bfrag(sW0, 64, lane, 1, 32), c1);
        c0 = relu8(c0); c1 = relu8(c1);
        __syncthreads();
        store_stage(S, lane, c0, c1);
        __syncthreads();
        v16h ah = load_afrag_lds(S, lane);

        // Layer 1: [16x32] x [32x32]
        c0 = bias_c(sB1, lane, 0); c1 = bias_c(sB1, lane, 1);
        c0 = wmma_f(ah, load_bfrag(sW1, 32, lane, 0, 0), c0);
        c1 = wmma_f(ah, load_bfrag(sW1, 32, lane, 1, 0), c1);
        c0 = relu8(c0); c1 = relu8(c1);
        __syncthreads();
        store_stage(S, lane, c0, c1);
        __syncthreads();
        ah = load_afrag_lds(S, lane);

        // Alpha head: [16x32] x [32x1->16] (kept in registers)
        v8f ca = bias_c(sBa, lane, 0);
        ca = wmma_f(ah, load_bfrag(sWa, 32, lane, 0, 0), ca);

        // Feature: [16x32] x [32x32] (no relu)
        c0 = bias_c(sBf, lane, 0); c1 = bias_c(sBf, lane, 1);
        c0 = wmma_f(ah, load_bfrag(sWf, 32, lane, 0, 0), c0);
        c1 = wmma_f(ah, load_bfrag(sWf, 32, lane, 1, 0), c1);
        __syncthreads();
        store_stage(S, lane, c0, c1);
        __syncthreads();
        v16h af = load_afrag_lds(S, lane);              // feature -> K 0..31

        // Direction layer: [16x59] x [59x32], K padded to 64
        c0 = bias_c(sBd, lane, 0); c1 = bias_c(sBd, lane, 1);
        c0 = wmma_f(af, load_bfrag(sWd, 64, lane, 0, 0),  c0);
        c0 = wmma_f(ad, load_bfrag(sWd, 64, lane, 0, 32), c0);
        c1 = wmma_f(af, load_bfrag(sWd, 64, lane, 1, 0),  c1);
        c1 = wmma_f(ad, load_bfrag(sWd, 64, lane, 1, 32), c1);
        c0 = relu8(c0); c1 = relu8(c1);
        __syncthreads();
        store_stage(S, lane, c0, c1);
        __syncthreads();
        ah = load_afrag_lds(S, lane);

        // RGB head: [16x32] x [32x3->16]
        v8f cr = bias_c(sBr, lane, 0);
        cr = wmma_f(ah, load_bfrag(sWr, 32, lane, 0, 0), cr);

        // ---- stage rgb(cols 0..2) + alpha(col 3) and write float4/point ----
        float* SF = (float*)S;                          // [16][4] f32 view
        __syncthreads();
        {
            const int n  = lane & 15;
            const int mh = (lane >> 4) << 3;
            if (n < 3) {
#pragma unroll
                for (int r = 0; r < 8; ++r) SF[(mh + r) * 4 + n] = cr[r];
            }
            if (n == 0) {
#pragma unroll
                for (int r = 0; r < 8; ++r) SF[(mh + r) * 4 + 3] = ca[r];
            }
        }
        __syncthreads();
        if (lane < 16) {
            const float4 o = ((const float4*)SF)[lane];
            *(float4*)(onet + (size_t)(m0 + lane) * 4) = o;
        }
    }
}

extern "C" void kernel_launch(void* const* d_in, const int* in_sizes, int n_in,
                              void* d_out, int out_size, void* d_ws, size_t ws_size,
                              hipStream_t stream) {
    (void)in_sizes; (void)n_in; (void)d_ws; (void)ws_size; (void)out_size;
    const float* xp  = (const float*)d_in[0];
    const float* W0p = (const float*)d_in[1];
    const float* b0p = (const float*)d_in[2];
    const float* W1p = (const float*)d_in[3];
    const float* b1p = (const float*)d_in[4];
    const float* Wap = (const float*)d_in[5];
    const float* bap = (const float*)d_in[6];
    const float* Wfp = (const float*)d_in[7];
    const float* bfp = (const float*)d_in[8];
    const float* Wdp = (const float*)d_in[9];
    const float* bdp = (const float*)d_in[10];
    const float* Wrp = (const float*)d_in[11];
    const float* brp = (const float*)d_in[12];
    float* outp = (float*)d_out;

    kilonerf_kernel<<<NNET, 256, 0, stream>>>(
        xp, W0p, b0p, W1p, b1p, Wap, bap, Wfp, bfp, Wdp, bdp, Wrp, brp, outp);
}
